// TASTF_70918499991820
// MI455X (gfx1250) — matrix operand order repaired and checked by reading, not compiled, not run on hardware
//
#include <hip/hip_runtime.h>
#include <math.h>

// Problem constants (from reference)
#define Tt   24
#define Nn   30000
#define NMm  6000
#define NPp  12000
#define NFf  12000
#define Dd   64
#define HIDh 32
#define DFFd 128
#define HORh 3

typedef __attribute__((ext_vector_type(16))) _Float16 v16h;
typedef __attribute__((ext_vector_type(8)))  float    v8f;

// ---------------------------------------------------------------------------
// WMMA helpers (CDNA5 v_wmma_f32_16x16x32_f16). One wave per 16-row tile.
// A 16x32 layout (ISA 7.12.2): lane m = lane&15; lanes<16 hold K {0..7,16..23},
// lanes>=16 hold K {8..15,24..31} -> per lane: two contiguous 8-float runs.
// B 32x16: n = lane&15; same K order as A (lanes<16 K {0..7,16..23}, lanes>=16 K {8..15,24..31})
// -> per lane: two contiguous 8-float runs of W row n (W is (Nout x K), out = x@W.T).
// C/D f32: lane n = lane&15, row m = vgpr + 8*(lane>=16).
// ---------------------------------------------------------------------------
__device__ __forceinline__ v16h pack16(const float4& x0, const float4& x1,
                                       const float4& y0, const float4& y1) {
  v16h a;
  a[0]  = (_Float16)x0.x; a[1]  = (_Float16)x0.y; a[2]  = (_Float16)x0.z; a[3]  = (_Float16)x0.w;
  a[4]  = (_Float16)x1.x; a[5]  = (_Float16)x1.y; a[6]  = (_Float16)x1.z; a[7]  = (_Float16)x1.w;
  a[8]  = (_Float16)y0.x; a[9]  = (_Float16)y0.y; a[10] = (_Float16)y0.z; a[11] = (_Float16)y0.w;
  a[12] = (_Float16)y1.x; a[13] = (_Float16)y1.y; a[14] = (_Float16)y1.z; a[15] = (_Float16)y1.w;
  return a;
}

__device__ __forceinline__ float4 scale4(float4 v, float s) {
  v.x *= s; v.y *= s; v.z *= s; v.w *= s; return v;
}

// A fragment from row-major f32 source (global or LDS), lda multiple of 4.
__device__ __forceinline__ v16h load_a_frag(const float* __restrict__ base, int lda, int kbase) {
  int lane = threadIdx.x & 31, m = lane & 15, hi = lane >> 4;
  const float* p = base + (size_t)m * lda + kbase + (hi << 3);
  const float4* p4 = (const float4*)p;
  const float4* q4 = (const float4*)(p + 16);
  return pack16(p4[0], p4[1], q4[0], q4[1]);
}

// Same but each row scaled by a per-row scalar (mean aggregation)
__device__ __forceinline__ v16h load_a_frag_scaled(const float* __restrict__ base, int lda,
                                                   int kbase, float s) {
  int lane = threadIdx.x & 31, m = lane & 15, hi = lane >> 4;
  const float* p = base + (size_t)m * lda + kbase + (hi << 3);
  const float4* p4 = (const float4*)p;
  const float4* q4 = (const float4*)(p + 16);
  return pack16(scale4(p4[0], s), scale4(p4[1], s), scale4(q4[0], s), scale4(q4[1], s));
}

// B fragment from weight W (Nout x K row-major); B[k][n] = W[nbase+n][kbase+k]
__device__ __forceinline__ v16h load_b_frag(const float* __restrict__ W, int ldw,
                                            int nbase, int kbase) {
  int lane = threadIdx.x & 31, n = lane & 15, hi = lane >> 4;
  const float4* p4 = (const float4*)(W + (size_t)(nbase + n) * ldw + kbase + (hi << 3));
  return pack16(p4[0], p4[1], p4[4], p4[5]);
}

__device__ __forceinline__ v8f wmma32(v16h a, v16h b, v8f c) {
  return __builtin_amdgcn_wmma_f32_16x16x32_f16(false, a, false, b, (short)0, c, false, false);
}

// ---------------------------------------------------------------------------
// Fused QKV: one wave per 16-row tile; X read ONCE; 24 WMMAs/wave.
// Epilogue staged through LDS -> linear float4 (b128) stores.
// ---------------------------------------------------------------------------
__global__ __launch_bounds__(32) void qkv_kernel(const float* __restrict__ X,
    const float* __restrict__ Wq, const float* __restrict__ bq,
    const float* __restrict__ Wk, const float* __restrict__ bk,
    const float* __restrict__ Wv, const float* __restrict__ bv,
    float* __restrict__ Q, float* __restrict__ K, float* __restrict__ V) {
  __shared__ __align__(16) float tile[16 * Dd];
  __builtin_prefetch(Wq, 0, 3);
  __builtin_prefetch(Wk, 0, 3);
  __builtin_prefetch(Wv, 0, 3);
  const float* At = X + (size_t)blockIdx.x * 16 * Dd;
  int lane = threadIdx.x & 31, n = lane & 15, hi = lane >> 4;
  v8f cq[4] = {}, ck[4] = {}, cv[4] = {};
#pragma unroll
  for (int kb = 0; kb < Dd; kb += 32) {
    v16h a = load_a_frag(At, Dd, kb);
#pragma unroll
    for (int nt = 0; nt < 4; ++nt) {
      cq[nt] = wmma32(a, load_b_frag(Wq, Dd, nt * 16, kb), cq[nt]);
      ck[nt] = wmma32(a, load_b_frag(Wk, Dd, nt * 16, kb), ck[nt]);
      cv[nt] = wmma32(a, load_b_frag(Wv, Dd, nt * 16, kb), cv[nt]);
    }
  }
  size_t obase = (size_t)blockIdx.x * 16 * Dd;
  const v8f* accs[3] = { cq, ck, cv };
  const float* biases[3] = { bq, bk, bv };
  float* outs[3] = { Q, K, V };
#pragma unroll
  for (int w = 0; w < 3; ++w) {
    __syncthreads();
    const v8f* c = accs[w];
#pragma unroll
    for (int r = 0; r < 8; ++r) {
      int row = r + 8 * hi;
#pragma unroll
      for (int nt = 0; nt < 4; ++nt) {
        int col = nt * 16 + n;
        tile[row * Dd + col] = c[nt][r] + biases[w][col];
      }
    }
    __syncthreads();
    const float4* src = (const float4*)tile;
    float4* dst = (float4*)(outs[w] + obase);
#pragma unroll
    for (int i = 0; i < 8; ++i) dst[lane + 32 * i] = src[lane + 32 * i];
  }
}

// ---------------------------------------------------------------------------
// FFN up (64 -> 128) + ReLU: one wave per 16-row tile; X read ONCE; 16 WMMAs.
// ---------------------------------------------------------------------------
__global__ __launch_bounds__(32) void ffn_up_kernel(const float* __restrict__ X,
                                                    const float* __restrict__ W1,
                                                    const float* __restrict__ b1,
                                                    float* __restrict__ Out) {
  __shared__ __align__(16) float tile[16 * DFFd];
  __builtin_prefetch(W1, 0, 3);
  const float* At = X + (size_t)blockIdx.x * 16 * Dd;
  int lane = threadIdx.x & 31, n = lane & 15, hi = lane >> 4;
  v8f c[8] = {};
#pragma unroll
  for (int kb = 0; kb < Dd; kb += 32) {
    v16h a = load_a_frag(At, Dd, kb);
#pragma unroll
    for (int nt = 0; nt < 8; ++nt)
      c[nt] = wmma32(a, load_b_frag(W1, Dd, nt * 16, kb), c[nt]);
  }
#pragma unroll
  for (int r = 0; r < 8; ++r) {
    int row = r + 8 * hi;
#pragma unroll
    for (int nt = 0; nt < 8; ++nt) {
      int col = nt * 16 + n;
      tile[row * DFFd + col] = fmaxf(c[nt][r] + b1[col], 0.0f);
    }
  }
  __syncthreads();
  const float4* src = (const float4*)tile;
  float4* dst = (float4*)(Out + (size_t)blockIdx.x * 16 * DFFd);
#pragma unroll
  for (int i = 0; i < 16; ++i) dst[lane + 32 * i] = src[lane + 32 * i];
}

// ---------------------------------------------------------------------------
// Fused GEMM(Nout=64) + bias + residual + LayerNorm, in-place on Xio.
// K is a template constant -> constant-stride addressing, fully unrolled.
// ---------------------------------------------------------------------------
template <int K>
__global__ __launch_bounds__(32) void gemm_res_ln_kernel(
    const float* __restrict__ A, const float* __restrict__ W, const float* __restrict__ bias,
    float* __restrict__ Xio, const float* __restrict__ g, const float* __restrict__ b) {
  __shared__ __align__(16) float tile[16 * Dd];
  __shared__ __align__(16) float rtile[16 * Dd];
  __builtin_prefetch(W, 0, 3);
  const float* At = A + (size_t)blockIdx.x * 16 * K;
  float* Xt = Xio + (size_t)blockIdx.x * 16 * Dd;
  int lane = threadIdx.x & 31, n = lane & 15, hi = lane >> 4;

  {  // coalesced residual stage-in (16x64 tile is one contiguous 4KB span)
    const float4* src = (const float4*)Xt;
    float4* dst = (float4*)rtile;
#pragma unroll
    for (int i = 0; i < 8; ++i) dst[lane + 32 * i] = src[lane + 32 * i];
  }
  v8f c[4] = {};
#pragma unroll
  for (int kb = 0; kb < K; kb += 32) {
    v16h a = load_a_frag(At, K, kb);
#pragma unroll
    for (int nt = 0; nt < 4; ++nt) c[nt] = wmma32(a, load_b_frag(W, K, nt * 16, kb), c[nt]);
  }
  __syncthreads();
#pragma unroll
  for (int r = 0; r < 8; ++r) {
    int row = r + 8 * hi;
#pragma unroll
    for (int nt = 0; nt < 4; ++nt) {
      int col = nt * 16 + n;
      tile[row * Dd + col] = c[nt][r] + bias[col] + rtile[row * Dd + col];
    }
  }
  __syncthreads();
  if (lane < 16) {  // per-row LayerNorm
    float mu = 0.f;
    for (int d = 0; d < Dd; ++d) mu += tile[lane * Dd + d];
    mu *= (1.0f / Dd);
    float var = 0.f;
    for (int d = 0; d < Dd; ++d) { float t = tile[lane * Dd + d] - mu; var += t * t; }
    var *= (1.0f / Dd);
    float rs = rsqrtf(var + 1e-5f);
    for (int d = 0; d < Dd; ++d)
      tile[lane * Dd + d] = (tile[lane * Dd + d] - mu) * rs * g[d] + b[d];
  }
  __syncthreads();
  {  // coalesced writeback
    const float4* src = (const float4*)tile;
    float4* dst = (float4*)Xt;
#pragma unroll
    for (int i = 0; i < 8; ++i) dst[lane + 32 * i] = src[lane + 32 * i];
  }
}

// ---------------------------------------------------------------------------
// GNN sparse aggregation (segment-sum via float atomics)
// ---------------------------------------------------------------------------
__global__ void cnt_kernel(const int* __restrict__ dst, int E, float* __restrict__ cnt) {
  int e = blockIdx.x * blockDim.x + threadIdx.x;
  if (e < E) atomicAdd(&cnt[dst[e]], 1.0f);
}

__global__ void scatter1_kernel(const int* __restrict__ src, const int* __restrict__ dst, int E,
                                const float* __restrict__ xs, float* __restrict__ agg) {
  int e = blockIdx.x * blockDim.x + threadIdx.x;
  if (e < E) atomicAdd(&agg[dst[e]], xs[src[e]]);
}

__global__ void scatter2_kernel(const int* __restrict__ src, const int* __restrict__ dst, int E,
                                const float* __restrict__ Hs, float* __restrict__ agg) {
  int tid = blockIdx.x * blockDim.x + threadIdx.x;
  int e = tid >> 5, j = tid & 31;
  if (e < E) atomicAdd(&agg[(size_t)dst[e] * HIDh + j], Hs[(size_t)src[e] * HIDh + j]);
}

// ---------------------------------------------------------------------------
// conv1 (din=1 -> HID=32) + ReLU. Relations: 0=mm 1=pp 2=ff 3=mp 4=mf.
// AGG/CNT scalar layout: [mm:NM][pp:NP][mp:NP][ff:NF][mf:NF]
// ---------------------------------------------------------------------------
struct Conv1Args {
  const float* xt;
  const float* agg;
  const float* cnt;
  const float* Wl[5];
  const float* bl[5];
  const float* Wr[5];
  float* H;
};

__global__ void conv1_kernel(Conv1Args a) {
  int tid = blockIdx.x * blockDim.x + threadIdx.x;
  if (tid >= Nn * HIDh) return;
  int n = tid >> 5, j = tid & 31;
  float x = a.xt[n];
  float h;
  if (n < NMm) {
    float m1 = a.agg[n] / fmaxf(a.cnt[n], 1.0f);
    h = m1 * a.Wl[0][j] + a.bl[0][j] + x * a.Wr[0][j];
  } else if (n < NMm + NPp) {
    int i = n - NMm;
    int o1 = NMm + i, o2 = NMm + NPp + i;
    float m1 = a.agg[o1] / fmaxf(a.cnt[o1], 1.0f);
    float m2 = a.agg[o2] / fmaxf(a.cnt[o2], 1.0f);
    h = m1 * a.Wl[1][j] + a.bl[1][j] + x * a.Wr[1][j]
      + m2 * a.Wl[3][j] + a.bl[3][j] + x * a.Wr[3][j];
  } else {
    int i = n - NMm - NPp;
    int o1 = NMm + 2 * NPp + i, o2 = NMm + 2 * NPp + NFf + i;
    float m1 = a.agg[o1] / fmaxf(a.cnt[o1], 1.0f);
    float m2 = a.agg[o2] / fmaxf(a.cnt[o2], 1.0f);
    h = m1 * a.Wl[2][j] + a.bl[2][j] + x * a.Wr[2][j]
      + m2 * a.Wl[4][j] + a.bl[4][j] + x * a.Wr[4][j];
  }
  a.H[(size_t)n * HIDh + j] = fmaxf(h, 0.0f);
}

// ---------------------------------------------------------------------------
// conv2 (WMMA, HID->D) + LayerNorm + ReLU + projection (WMMA) + PE epilogue.
// One wave per 16 dst nodes; up to 24 v_wmma per wave; linear b128 writeback.
// ---------------------------------------------------------------------------
__global__ __launch_bounds__(32) void conv2_ln_proj_kernel(
    const float* __restrict__ Hdst,
    const float* __restrict__ agg1, const float* __restrict__ cnt1,
    const float* __restrict__ Wl1, const float* __restrict__ bl1, const float* __restrict__ Wr1,
    int has2,
    const float* __restrict__ agg2, const float* __restrict__ cnt2,
    const float* __restrict__ Wl2, const float* __restrict__ bl2, const float* __restrict__ Wr2,
    const float* __restrict__ ng, const float* __restrict__ nb,
    const float* __restrict__ Wp, const float* __restrict__ bp,
    float* __restrict__ Xout, int tstep) {
  __shared__ __align__(16) float tile[16 * Dd];
  int lane = threadIdx.x & 31;
  int m = lane & 15, hi = lane >> 4, n = m;
  int node0 = blockIdx.x * 16;
  v8f c[4] = {};

  {  // mean1 @ Wl1^T
    float s = 1.0f / fmaxf(cnt1[node0 + m], 1.0f);
    v16h a = load_a_frag_scaled(agg1 + (size_t)node0 * HIDh, HIDh, 0, s);
#pragma unroll
    for (int nt = 0; nt < 4; ++nt) c[nt] = wmma32(a, load_b_frag(Wl1, HIDh, nt * 16, 0), c[nt]);
  }
  {  // h_dst @ Wr1^T
    v16h a = load_a_frag(Hdst + (size_t)node0 * HIDh, HIDh, 0);
#pragma unroll
    for (int nt = 0; nt < 4; ++nt) c[nt] = wmma32(a, load_b_frag(Wr1, HIDh, nt * 16, 0), c[nt]);
  }
  if (has2) {  // second relation into the same dst type
    float s = 1.0f / fmaxf(cnt2[node0 + m], 1.0f);
    v16h a = load_a_frag_scaled(agg2 + (size_t)node0 * HIDh, HIDh, 0, s);
#pragma unroll
    for (int nt = 0; nt < 4; ++nt) c[nt] = wmma32(a, load_b_frag(Wl2, HIDh, nt * 16, 0), c[nt]);
    v16h a2 = load_a_frag(Hdst + (size_t)node0 * HIDh, HIDh, 0);
#pragma unroll
    for (int nt = 0; nt < 4; ++nt) c[nt] = wmma32(a2, load_b_frag(Wr2, HIDh, nt * 16, 0), c[nt]);
  }

  // bias + stash pre-LN rows in LDS
#pragma unroll
  for (int r = 0; r < 8; ++r) {
    int row = r + 8 * hi;
#pragma unroll
    for (int nt = 0; nt < 4; ++nt) {
      int col = nt * 16 + n;
      tile[row * Dd + col] = c[nt][r] + bl1[col] + (has2 ? bl2[col] : 0.0f);
    }
  }
  __syncthreads();
  if (lane < 16) {  // per-row LayerNorm + ReLU
    float mu = 0.f;
    for (int d = 0; d < Dd; ++d) mu += tile[lane * Dd + d];
    mu *= (1.0f / Dd);
    float var = 0.f;
    for (int d = 0; d < Dd; ++d) { float t = tile[lane * Dd + d] - mu; var += t * t; }
    var *= (1.0f / Dd);
    float rs = rsqrtf(var + 1e-5f);
    for (int d = 0; d < Dd; ++d) {
      float v = (tile[lane * Dd + d] - mu) * rs * ng[d] + nb[d];
      tile[lane * Dd + d] = fmaxf(v, 0.0f);
    }
  }
  __syncthreads();
  // projection (64 -> 64)
  v8f c2[4] = {};
#pragma unroll
  for (int kb = 0; kb < Dd; kb += 32) {
    v16h a = load_a_frag(tile, Dd, kb);
#pragma unroll
    for (int nt = 0; nt < 4; ++nt) c2[nt] = wmma32(a, load_b_frag(Wp, Dd, nt * 16, kb), c2[nt]);
  }
  // positional-encoding epilogue: pe(t,col)
  float pe[4];
#pragma unroll
  for (int nt = 0; nt < 4; ++nt) {
    int col = nt * 16 + n;
    float div = expf((float)(col & ~1) * (-9.210340372f / 64.0f));  // exp(-ln(1e4)*2i/D)
    float ang = (float)tstep * div;
    pe[nt] = (col & 1) ? cosf(ang) : sinf(ang);
  }
  __syncthreads();  // done reading tile as A; safe to overwrite
#pragma unroll
  for (int r = 0; r < 8; ++r) {
    int row = r + 8 * hi;
#pragma unroll
    for (int nt = 0; nt < 4; ++nt)
      tile[row * Dd + nt * 16 + n] = c2[nt][r] + bp[nt * 16 + n] + pe[nt];
  }
  __syncthreads();
  {  // coalesced writeback (rows are contiguous in Xout)
    const float4* src = (const float4*)tile;
    float4* dst = (float4*)(Xout + (size_t)node0 * Dd);
#pragma unroll
    for (int i = 0; i < 8; ++i) dst[lane + 32 * i] = src[lane + 32 * i];
  }
}

// ---------------------------------------------------------------------------
// Attention over T=24 per node (4 heads x dh=16). Block = 1 node, 96 threads.
// Q/K/V rows staged in LDS via float4; output written back into Q buffer.
// ---------------------------------------------------------------------------
__global__ __launch_bounds__(96) void attn_kernel(float* __restrict__ Q,
                                                  const float* __restrict__ Kb,
                                                  const float* __restrict__ V) {
  __shared__ __align__(16) float lq[Tt * Dd];
  __shared__ __align__(16) float lk[Tt * Dd];
  __shared__ __align__(16) float lv[Tt * Dd];
  int node = blockIdx.x;
  for (int f = threadIdx.x; f < Tt * (Dd / 4); f += 96) {
    int t = f >> 4, dq = f & 15;
    size_t g4 = ((size_t)t * Nn + node) * (Dd / 4) + dq;
    ((float4*)lq)[f] = ((const float4*)Q)[g4];
    ((float4*)lk)[f] = ((const float4*)Kb)[g4];
    ((float4*)lv)[f] = ((const float4*)V)[g4];
  }
  __syncthreads();
  int h = threadIdx.x / Tt;  // 0..3
  int t = threadIdx.x % Tt;  // 0..23
  float sc[Tt];
  float mx = -1e30f;
  const float* qr = lq + t * Dd + h * 16;
  for (int u = 0; u < Tt; ++u) {
    const float* kr = lk + u * Dd + h * 16;
    float s = 0.f;
#pragma unroll
    for (int d = 0; d < 16; ++d) s += qr[d] * kr[d];
    s *= 0.25f;  // 1/sqrt(16)
    sc[u] = s;
    mx = fmaxf(mx, s);
  }
  float sum = 0.f;
  for (int u = 0; u < Tt; ++u) { sc[u] = expf(sc[u] - mx); sum += sc[u]; }
  float inv = 1.0f / sum;
  float o[16];
#pragma unroll
  for (int d = 0; d < 16; ++d) o[d] = 0.f;
  for (int u = 0; u < Tt; ++u) {
    float w = sc[u] * inv;
    const float* vr = lv + u * Dd + h * 16;
#pragma unroll
    for (int d = 0; d < 16; ++d) o[d] += w * vr[d];
  }
  float4* out4 = (float4*)Q;
  size_t g4 = ((size_t)t * Nn + node) * (Dd / 4) + h * 4;
  out4[g4 + 0] = make_float4(o[0], o[1], o[2], o[3]);
  out4[g4 + 1] = make_float4(o[4], o[5], o[6], o[7]);
  out4[g4 + 2] = make_float4(o[8], o[9], o[10], o[11]);
  out4[g4 + 3] = make_float4(o[12], o[13], o[14], o[15]);
}

// Head: out[h*N + n] = final[n] . W[h] + b[h]   (output (1, HOR, N))
__global__ void head_kernel(const float* __restrict__ Xlast, const float* __restrict__ W,
                            const float* __restrict__ b, float* __restrict__ out) {
  int n = blockIdx.x * blockDim.x + threadIdx.x;
  if (n >= Nn) return;
  const float* x = Xlast + (size_t)n * Dd;
  for (int h = 0; h < HORh; ++h) {
    float acc = b[h];
    for (int d = 0; d < Dd; ++d) acc += x[d] * W[h * Dd + d];
    out[h * Nn + n] = acc;
  }
}

// ---------------------------------------------------------------------------
// Host orchestration. Input leaves follow setup_inputs() dict insertion order:
// 0:x_seq 1:e_mm 2:e_pp 3:e_ff 4:e_mp 5:e_mf
// 6..20  conv1 {mm,pp,ff,mp,mf} x {Wl,bl,Wr}
// 21..35 conv2 {mm,pp,ff,mp,mf} x {Wl,bl,Wr}
// 36:norm_g 37:norm_b  38..43 proj {macro,pico,femto} x {W,b}
// 44..59 tf layer0 {Wq,bq,Wk,bk,Wv,bv,Wo,bo,W1,b1,W2,b2,g1,be1,g2,be2}
// 60..75 tf layer1 (same)   76:head_W 77:head_b
// ---------------------------------------------------------------------------
extern "C" void kernel_launch(void* const* d_in, const int* in_sizes, int n_in,
                              void* d_out, int out_size, void* d_ws, size_t ws_size,
                              hipStream_t stream) {
  (void)n_in; (void)out_size; (void)ws_size;
#define F(i) ((const float*)d_in[(i)])
  const float* x_seq = F(0);
  const int* e_mm = (const int*)d_in[1];
  const int* e_pp = (const int*)d_in[2];
  const int* e_ff = (const int*)d_in[3];
  const int* e_mp = (const int*)d_in[4];
  const int* e_mf = (const int*)d_in[5];
  const int Emm = in_sizes[1] / 2, Epp = in_sizes[2] / 2, Eff = in_sizes[3] / 2,
            Emp = in_sizes[4] / 2, Emf = in_sizes[5] / 2;

  // Workspace carve-up
  const size_t TND = (size_t)Tt * Nn * Dd;           // 46.08M floats
  const int AGGN = NMm + 2 * NPp + 2 * NFf;          // 54000
  float* X    = (float*)d_ws;                        // (T,N,D) sequence / transformer state
  float* Qb   = X + TND;                             // Q (also attn output, FFN hidden lo)
  float* Kb   = Qb + TND;                             // K (also FFN hidden hi)
  float* Vb   = Kb + TND;                            // V
  float* FFb  = Qb;                                  // (T*N,128) aliases Qb+Kb (2*TND)
  float* CNT  = Vb + TND;                            // degree counts      [AGGN]
  float* AGG1 = CNT + AGGN;                          // conv1 scalar agg   [AGGN]
  float* AGG2 = AGG1 + AGGN;                         // conv2 vector agg   [AGGN*32]
  float* Hb   = AGG2 + (size_t)AGGN * HIDh;          // conv1 output (N,32)

  // ---- degree counts (t-independent) ----
  hipMemsetAsync(CNT, 0, (size_t)AGGN * sizeof(float), stream);
  cnt_kernel<<<(Emm + 255) / 256, 256, 0, stream>>>(e_mm + Emm, Emm, CNT);
  cnt_kernel<<<(Epp + 255) / 256, 256, 0, stream>>>(e_pp + Epp, Epp, CNT + NMm);
  cnt_kernel<<<(Emp + 255) / 256, 256, 0, stream>>>(e_mp + Emp, Emp, CNT + NMm + NPp);
  cnt_kernel<<<(Eff + 255) / 256, 256, 0, stream>>>(e_ff + Eff, Eff, CNT + NMm + 2 * NPp);
  cnt_kernel<<<(Emf + 255) / 256, 256, 0, stream>>>(e_mf + Emf, Emf, CNT + NMm + 2 * NPp + NFf);

  // ---- per-timestep GNN ----
  for (int t = 0; t < Tt; ++t) {
    const float* xt = x_seq + (size_t)t * Nn;
    hipMemsetAsync(AGG1, 0, (size_t)AGGN * sizeof(float), stream);
    scatter1_kernel<<<(Emm + 255) / 256, 256, 0, stream>>>(e_mm, e_mm + Emm, Emm, xt, AGG1);
    scatter1_kernel<<<(Epp + 255) / 256, 256, 0, stream>>>(e_pp, e_pp + Epp, Epp, xt + NMm, AGG1 + NMm);
    scatter1_kernel<<<(Emp + 255) / 256, 256, 0, stream>>>(e_mp, e_mp + Emp, Emp, xt, AGG1 + NMm + NPp);
    scatter1_kernel<<<(Eff + 255) / 256, 256, 0, stream>>>(e_ff, e_ff + Eff, Eff, xt + NMm + NPp, AGG1 + NMm + 2 * NPp);
    scatter1_kernel<<<(Emf + 255) / 256, 256, 0, stream>>>(e_mf, e_mf + Emf, Emf, xt, AGG1 + NMm + 2 * NPp + NFf);

    Conv1Args ca = { xt, AGG1, CNT,
                     { F(6), F(9), F(12), F(15), F(18) },
                     { F(7), F(10), F(13), F(16), F(19) },
                     { F(8), F(11), F(14), F(17), F(20) },
                     Hb };
    conv1_kernel<<<(Nn * HIDh + 255) / 256, 256, 0, stream>>>(ca);

    hipMemsetAsync(AGG2, 0, (size_t)AGGN * HIDh * sizeof(float), stream);
    scatter2_kernel<<<((size_t)Emm * 32 + 255) / 256, 256, 0, stream>>>(e_mm, e_mm + Emm, Emm, Hb, AGG2);
    scatter2_kernel<<<((size_t)Epp * 32 + 255) / 256, 256, 0, stream>>>(e_pp, e_pp + Epp, Epp, Hb + (size_t)NMm * HIDh, AGG2 + (size_t)NMm * HIDh);
    scatter2_kernel<<<((size_t)Emp * 32 + 255) / 256, 256, 0, stream>>>(e_mp, e_mp + Emp, Emp, Hb, AGG2 + (size_t)(NMm + NPp) * HIDh);
    scatter2_kernel<<<((size_t)Eff * 32 + 255) / 256, 256, 0, stream>>>(e_ff, e_ff + Eff, Eff, Hb + (size_t)(NMm + NPp) * HIDh, AGG2 + (size_t)(NMm + 2 * NPp) * HIDh);
    scatter2_kernel<<<((size_t)Emf * 32 + 255) / 256, 256, 0, stream>>>(e_mf, e_mf + Emf, Emf, Hb, AGG2 + (size_t)(NMm + 2 * NPp + NFf) * HIDh);

    float* Xt = X + (size_t)t * Nn * Dd;
    // macro: conv2 mm + proj macro (+PE)
    conv2_ln_proj_kernel<<<NMm / 16, 32, 0, stream>>>(
        Hb, AGG2, CNT, F(21), F(22), F(23),
        0, nullptr, nullptr, nullptr, nullptr, nullptr,
        F(36), F(37), F(38), F(39), Xt, t);
    // pico: conv2 pp + mp + proj pico (+PE)
    conv2_ln_proj_kernel<<<NPp / 16, 32, 0, stream>>>(
        Hb + (size_t)NMm * HIDh, AGG2 + (size_t)NMm * HIDh, CNT + NMm, F(24), F(25), F(26),
        1, AGG2 + (size_t)(NMm + NPp) * HIDh, CNT + NMm + NPp, F(30), F(31), F(32),
        F(36), F(37), F(40), F(41), Xt + (size_t)NMm * Dd, t);
    // femto: conv2 ff + mf + proj femto (+PE)
    conv2_ln_proj_kernel<<<NFf / 16, 32, 0, stream>>>(
        Hb + (size_t)(NMm + NPp) * HIDh, AGG2 + (size_t)(NMm + 2 * NPp) * HIDh, CNT + NMm + 2 * NPp,
        F(27), F(28), F(29),
        1, AGG2 + (size_t)(NMm + 2 * NPp + NFf) * HIDh, CNT + NMm + 2 * NPp + NFf, F(33), F(34), F(35),
        F(36), F(37), F(42), F(43), Xt + (size_t)(NMm + NPp) * Dd, t);
  }

  // ---- transformer layers (all dense matmuls via WMMA, LN fused) ----
  const int Mrows = Tt * Nn;  // 720000, divisible by 16
  const int RT = Mrows / 16;  // 45000 row tiles
  for (int L = 0; L < 2; ++L) {
    int p = 44 + 16 * L;
    qkv_kernel<<<RT, 32, 0, stream>>>(X, F(p + 0), F(p + 1), F(p + 2), F(p + 3),
                                      F(p + 4), F(p + 5), Qb, Kb, Vb);
    attn_kernel<<<Nn, 96, 0, stream>>>(Qb, Kb, Vb);                                  // O -> Qb
    gemm_res_ln_kernel<Dd><<<RT, 32, 0, stream>>>(Qb, F(p + 6), F(p + 7), X,
                                                  F(p + 12), F(p + 13));             // +Wo, +res, LN1
    ffn_up_kernel<<<RT, 32, 0, stream>>>(X, F(p + 8), F(p + 9), FFb);                // FFN up + relu
    gemm_res_ln_kernel<DFFd><<<RT, 32, 0, stream>>>(FFb, F(p + 10), F(p + 11), X,
                                                    F(p + 14), F(p + 15));           // +res, LN2
  }

  // ---- head on last timestep ----
  head_kernel<<<(Nn + 255) / 256, 256, 0, stream>>>(
      X + (size_t)(Tt - 1) * Nn * Dd, F(76), F(77), (float*)d_out);
#undef F
}
